// S6_46368466927945
// MI455X (gfx1250) — matrix operand order repaired and checked by reading, not compiled, not run on hardware
//
#include <hip/hip_runtime.h>
#include <hip/hip_bf16.h>
#include <math.h>

// ---------------- problem constants ----------------
#define BB   2
#define LL   4096
#define DD   256              // d_model == K of every GEMM
#define NS   16               // state size
#define MM   (BB * LL)        // 8192 rows
#define CH   32               // scan chunks
#define CLEN (LL / CH)        // 128 steps per chunk

typedef __attribute__((ext_vector_type(16))) __bf16 v16bf;
typedef __attribute__((ext_vector_type(8)))  __bf16 v8bf;
typedef __attribute__((ext_vector_type(8)))  float  v8f;
typedef __attribute__((ext_vector_type(4)))  int    v4i;

#if defined(__HIP_DEVICE_COMPILE__) && __has_builtin(__builtin_amdgcn_global_load_async_to_lds_b128)
#define USE_ASYNC_LDS 1
typedef __attribute__((address_space(1))) v4i* as1_v4i_ptr;
typedef __attribute__((address_space(3))) v4i* as3_v4i_ptr;
#endif

__device__ __forceinline__ void wait_async_zero() {
#if defined(__HIP_DEVICE_COMPILE__)
#if __has_builtin(__builtin_amdgcn_s_wait_asynccnt)
  __builtin_amdgcn_s_wait_asynccnt(0);
#else
  asm volatile("s_wait_asynccnt 0" ::: "memory");
#endif
#endif
}

// ---------------- RMSNorm (one row of 256 per block) ----------------
__global__ void rmsnorm_k(const float* __restrict__ in, float* __restrict__ outf,
                          __bf16* __restrict__ outb) {
  __shared__ float red[DD];
  const int row = blockIdx.x;
  const int tid = threadIdx.x;
  const float v = in[row * DD + tid];
  red[tid] = v * v;
  __syncthreads();
  for (int s = DD / 2; s > 0; s >>= 1) {
    if (tid < s) red[tid] += red[tid + s];
    __syncthreads();
  }
  const float scale = rsqrtf(red[0] * (1.0f / DD) + 1.1920929e-7f);
  const float o = v * scale;
  if (outf) outf[row * DD + tid] = o;
  if (outb) outb[row * DD + tid] = (__bf16)o;
}

// ---------------- fp32 [K,Nsrc] -> bf16 transposed [colOff+Nsrc, K] ----------------
__global__ void wconv_k(const float* __restrict__ src, __bf16* __restrict__ dstT,
                        int Nsrc, int colOff) {
  const int i = blockIdx.x * blockDim.x + threadIdx.x;
  if (i >= DD * Nsrc) return;
  const int k = i / Nsrc;
  const int n = i % Nsrc;
  dstT[(colOff + n) * DD + k] = (__bf16)src[k * Nsrc + n];
}

// ---------------- WMMA GEMM: out[M,NN] = A[M,256] * WtT[NN,256]^T ------------
// One wave computes a 16x32 tile (two 16x16 WMMA tiles sharing one A fragment).
// Block stages the 16x256 A tile in LDS via gfx1250 async global->LDS.
// mode: 0 = plain, 1 = +bias, 2 = softplus(acc + bias)  (fused dt epilogue)
__global__ void gemm_wmma(const __bf16* __restrict__ A, const __bf16* __restrict__ Wt,
                          float* __restrict__ out, const float* __restrict__ bias,
                          int NN, int mode) {
  __shared__ __bf16 ldsA[16 * DD];           // 8 KB
  const int tid  = threadIdx.x;
  const int lane = tid & 31;
  const int wave = tid >> 5;
  const int mBase   = blockIdx.x * 16;
  const int colBase = wave * 32;             // each wave: 32 output columns

  // ---- stage A tile (16 rows x 256 K, bf16) into LDS ----
  const __bf16* gA = A + (size_t)mBase * DD;
  const int nChunks = (16 * DD) / 8;         // 512 x 16B
  for (int c = tid; c < nChunks; c += blockDim.x) {
#ifdef USE_ASYNC_LDS
    __builtin_amdgcn_global_load_async_to_lds_b128(
        (as1_v4i_ptr)(gA + c * 8),
        (as3_v4i_ptr)(ldsA + c * 8), 0, 0);
#else
    *(v8bf*)(ldsA + c * 8) = *(const v8bf*)(gA + c * 8);
#endif
  }
#ifdef USE_ASYNC_LDS
  wait_async_zero();
#endif
  __syncthreads();

  const int r    = lane & 15;                // A row within tile / B column within tile
  const int hsel = lane >> 4;
  const __bf16* wcol0 = Wt + (size_t)(colBase + r) * DD;
  const __bf16* wcol1 = Wt + (size_t)(colBase + 16 + r) * DD;

  union Frag { v16bf v; v8bf h[2]; };
  v8f acc0 = {};
  v8f acc1 = {};
  #pragma unroll
  for (int k0 = 0; k0 < DD; k0 += 32) {
    Frag af, bf0, bf1;
    // ISA A layout (16-bit 16x32): lanes 0-15 hold K[0..7],K[16..23]; lanes 16-31 K[8..15],K[24..31]
    af.h[0] = *(const v8bf*)&ldsA[r * DD + k0 + hsel * 8];
    af.h[1] = *(const v8bf*)&ldsA[r * DD + k0 + 16 + hsel * 8];
    // ISA B layout (32x16): lane holds one N column; lanes 0-15 K[0..7],K[16..23]; lanes 16-31 K[8..15],K[24..31]
    const int kb = k0 + hsel * 8;
    bf0.h[0] = *(const v8bf*)&wcol0[kb];
    bf0.h[1] = *(const v8bf*)&wcol0[kb + 16];
    bf1.h[0] = *(const v8bf*)&wcol1[kb];
    bf1.h[1] = *(const v8bf*)&wcol1[kb + 16];
    acc0 = __builtin_amdgcn_wmma_f32_16x16x32_bf16(false, af.v, false, bf0.v,
                                                   (short)0, acc0, false, false);
    acc1 = __builtin_amdgcn_wmma_f32_16x16x32_bf16(false, af.v, false, bf1.v,
                                                   (short)0, acc1, false, false);
  }

  // C/D layout: VGPR i -> M = i + hsel*8 ; N = lane&15
  #pragma unroll
  for (int t = 0; t < 2; ++t) {
    const int col = colBase + t * 16 + r;
    const float bv = (mode >= 1) ? bias[col] : 0.0f;
    #pragma unroll
    for (int i = 0; i < 8; ++i) {
      const int row = mBase + hsel * 8 + i;
      float v = (t == 0 ? acc0[i] : acc1[i]) + bv;
      if (mode == 2) v = (v > 20.0f) ? v : log1pf(__expf(v));   // softplus
      out[(size_t)row * NN + col] = v;
    }
  }
}

// ---------------- scan pass 1: per-chunk aggregates (prodA, h_end | h0=0) -------------
// One wave per (b, d-pair, chunk). lanes 0-15 -> n of d0, lanes 16-31 -> n of d1.
__global__ void scan_pass1(const float* __restrict__ dt, const float* __restrict__ xn,
                           const float* __restrict__ BC, const float* __restrict__ Amat,
                           float* __restrict__ aprod, float* __restrict__ hend) {
  const int blk = blockIdx.x;
  const int c  = blk % CH;
  const int dp = (blk / CH) % (DD / 2);
  const int b  = blk / (CH * (DD / 2));
  const int lane = threadIdx.x;
  const int n = lane & 15;
  const int d = dp * 2 + (lane >> 4);
  const float Ad = Amat[d * NS + n];
  float ap = 1.0f, h = 0.0f;
  const int base = b * LL + c * CLEN;
  for (int s = 0; s < CLEN; ++s) {
    const int l = base + s;
    const float dts = dt[l * DD + d];
    const float xs  = xn[l * DD + d];
    const float Bv  = BC[l * 32 + n];
    const float a   = __expf(dts * Ad);
    h = a * h + dts * Bv * xs;
    ap *= a;
  }
  const int idx = ((b * CH + c) * DD + d) * NS + n;
  aprod[idx] = ap;
  hend[idx]  = h;
}

// ---------------- scan prefix across the 32 chunk aggregates ----------------
__global__ void scan_prefix(const float* __restrict__ aprod, const float* __restrict__ hend,
                            float* __restrict__ hinit) {
  const int blk = blockIdx.x;                // b * (DD/2)
  const int dp = blk % (DD / 2);
  const int b  = blk / (DD / 2);
  const int lane = threadIdx.x;
  const int n = lane & 15;
  const int d = dp * 2 + (lane >> 4);
  float h = 0.0f;
  for (int c = 0; c < CH; ++c) {
    const int idx = ((b * CH + c) * DD + d) * NS + n;
    hinit[idx] = h;
    h = aprod[idx] * h + hend[idx];
  }
}

// ---------------- scan pass 2: replay with h_init, contract N, +skip, GELU, ->bf16 ----
__global__ void scan_pass2(const float* __restrict__ dt, const float* __restrict__ xn,
                           const float* __restrict__ BC, const float* __restrict__ Amat,
                           const float* __restrict__ Dskip, const float* __restrict__ hinit,
                           __bf16* __restrict__ act) {
  const int blk = blockIdx.x;
  const int c  = blk % CH;
  const int dp = (blk / CH) % (DD / 2);
  const int b  = blk / (CH * (DD / 2));
  const int lane = threadIdx.x;
  const int n = lane & 15;
  const int d = dp * 2 + (lane >> 4);
  const float Ad = Amat[d * NS + n];
  const float Dd = Dskip[d];
  float h = hinit[((b * CH + c) * DD + d) * NS + n];
  const int base = b * LL + c * CLEN;
  for (int s = 0; s < CLEN; ++s) {
    const int l = base + s;
    const float dts = dt[l * DD + d];
    const float xs  = xn[l * DD + d];
    const float Bv  = BC[l * 32 + n];
    const float Cv  = BC[l * 32 + 16 + n];
    const float a   = __expf(dts * Ad);
    h = a * h + dts * Bv * xs;
    // contract over N within each 16-lane half (wave32 shuffles)
    float r = h * Cv;
    r += __shfl_xor(r, 1);
    r += __shfl_xor(r, 2);
    r += __shfl_xor(r, 4);
    r += __shfl_xor(r, 8);
    if (n == 0) {
      const float y = r + Dd * xs;
      const float t = 0.7978845608028654f * (y + 0.044715f * y * y * y);
      const float g = 0.5f * y * (1.0f + tanhf(t));
      act[l * DD + d] = (__bf16)g;
    }
  }
}

// ---------------- host-side orchestration ----------------
extern "C" void kernel_launch(void* const* d_in, const int* in_sizes, int n_in,
                              void* d_out, int out_size, void* d_ws, size_t ws_size,
                              hipStream_t stream) {
  (void)in_sizes; (void)n_in; (void)out_size; (void)ws_size;
  const float* x   = (const float*)d_in[0];
  const float* A   = (const float*)d_in[1];
  const float* Wd  = (const float*)d_in[2];
  const float* bd  = (const float*)d_in[3];
  const float* WB  = (const float*)d_in[4];
  const float* WC  = (const float*)d_in[5];
  const float* Dsk = (const float*)d_in[6];
  const float* Wm  = (const float*)d_in[7];
  const float* bm  = (const float*)d_in[8];
  float* out = (float*)d_out;

  char* p = (char*)d_ws;
  auto carve = [&](size_t bytes) -> void* {
    void* r = (void*)p;
    p += (bytes + 255) & ~(size_t)255;
    return r;
  };
  float*  xn_f  = (float*) carve((size_t)MM * DD * 4);
  __bf16* xn_b  = (__bf16*)carve((size_t)MM * DD * 2);
  __bf16* act_b = (__bf16*)carve((size_t)MM * DD * 2);
  float*  dtbuf = (float*) carve((size_t)MM * DD * 4);
  float*  BC    = (float*) carve((size_t)MM * 32 * 4);
  __bf16* WtD   = (__bf16*)carve((size_t)DD * DD * 2);
  __bf16* WtBC  = (__bf16*)carve((size_t)DD * 32 * 2);
  __bf16* WtM   = (__bf16*)carve((size_t)DD * DD * 2);
  float*  aprod = (float*) carve((size_t)BB * CH * DD * NS * 4);
  float*  hend  = (float*) carve((size_t)BB * CH * DD * NS * 4);
  float*  hinit = (float*) carve((size_t)BB * CH * DD * NS * 4);
  float*  xcur  = (float*) carve((size_t)MM * DD * 4);

  const float* xin = x;
  for (int l = 0; l < 2; ++l) {
    rmsnorm_k<<<MM, DD, 0, stream>>>(xin, xn_f, xn_b);

    wconv_k<<<(DD * DD + 255) / 256, 256, 0, stream>>>(Wd + (size_t)l * DD * DD, WtD, DD, 0);
    wconv_k<<<(DD * NS + 255) / 256, 256, 0, stream>>>(WB + (size_t)l * DD * NS, WtBC, NS, 0);
    wconv_k<<<(DD * NS + 255) / 256, 256, 0, stream>>>(WC + (size_t)l * DD * NS, WtBC, NS, 16);
    wconv_k<<<(DD * DD + 255) / 256, 256, 0, stream>>>(Wm + (size_t)l * DD * DD, WtM, DD, 0);

    // dt = softplus(xn @ W_delta + b_delta)   (fused epilogue, mode 2)
    gemm_wmma<<<MM / 16, 256, 0, stream>>>(xn_b, WtD, dtbuf, bd + (size_t)l * DD, DD, 2);
    // [Bm | Cm] = xn @ [W_B | W_C]            (one wave covers all 32 columns)
    gemm_wmma<<<MM / 16, 32, 0, stream>>>(xn_b, WtBC, BC, nullptr, 32, 0);

    scan_pass1<<<BB * (DD / 2) * CH, 32, 0, stream>>>(dtbuf, xn_f, BC,
                                                      A + (size_t)l * DD * NS, aprod, hend);
    scan_prefix<<<BB * (DD / 2), 32, 0, stream>>>(aprod, hend, hinit);
    scan_pass2<<<BB * (DD / 2) * CH, 32, 0, stream>>>(dtbuf, xn_f, BC,
                                                      A + (size_t)l * DD * NS,
                                                      Dsk + (size_t)l * DD, hinit, act_b);

    // x_next = gelu(y) @ W_mix + b_mix        (mode 1)
    gemm_wmma<<<MM / 16, 256, 0, stream>>>(act_b, WtM, xcur, bm + (size_t)l * DD, DD, 1);
    xin = xcur;
  }
  rmsnorm_k<<<MM, DD, 0, stream>>>(xin, out, nullptr);
}
